// TopologicalEncoder_31808527794372
// MI455X (gfx1250) — matrix operand order repaired and checked by reading, not compiled, not run on hardware
//
#include <hip/hip_runtime.h>
#include <hip/hip_bf16.h>

typedef __attribute__((ext_vector_type(16))) _Float16 v16h;
typedef __attribute__((ext_vector_type(8)))  _Float16 v8h;
typedef __attribute__((ext_vector_type(8)))  float    v8f;

#define BATCH 8
#define NPTS  2048
#define DIN   32
#define HID   64
#define FDIM  37      // structural feature dim with selection weight
#define ADIM  36      // structural feature dim without selection weight
#define KLIFT 16
#define DMODEL 256
#define KSEL  8.0f
#define KEFF  16      // min(N, MAX_PROXY)
#define CHUNK 256     // columns staged per TDM transfer
#define NCHUNK (NPTS / CHUNK)

// ---------------------------------------------------------------------------
// Kernel 1: per-point saliency MLP (32->64->1), plus f16 copy of x and sum x^2
// ---------------------------------------------------------------------------
__global__ __launch_bounds__(256) void k_saliency(
    const float* __restrict__ x, const float* __restrict__ W1,
    const float* __restrict__ b1, const float* __restrict__ W2,
    const float* __restrict__ b2,
    _Float16* __restrict__ xh, float* __restrict__ sq, float* __restrict__ sal) {
  __shared__ float sW1[DIN * HID];
  __shared__ float sb1[HID];
  __shared__ float sW2[HID];
  int tid = threadIdx.x;
  for (int i = tid; i < DIN * HID; i += 256) sW1[i] = W1[i];
  if (tid < HID) { sb1[tid] = b1[tid]; sW2[tid] = W2[tid]; }
  __syncthreads();

  int n = blockIdx.x * 256 + tid;                 // global point id (B*N)
  const float* xr = x + (size_t)n * DIN;
  float xl[DIN];
  float ss = 0.f;
#pragma unroll
  for (int d = 0; d < DIN; ++d) { xl[d] = xr[d]; ss += xl[d] * xl[d]; }
  sq[n] = ss;

  // packed f16 copy (16 dword stores per point)
  unsigned int* xo = (unsigned int*)(xh + (size_t)n * DIN);
#pragma unroll
  for (int i = 0; i < DIN / 2; ++i) {
    union { _Float16 h[2]; unsigned int u; } p;
    p.h[0] = (_Float16)xl[2 * i];
    p.h[1] = (_Float16)xl[2 * i + 1];
    xo[i] = p.u;
  }

  float s = b2[0];
#pragma unroll 4
  for (int hI = 0; hI < HID; ++hI) {
    float acc = sb1[hI];
#pragma unroll
    for (int d = 0; d < DIN; ++d) acc += xl[d] * sW1[d * HID + hI];
    acc = fmaxf(acc, 0.f);
    s += acc * sW2[hI];
  }
  sal[n] = s;
}

// ---------------------------------------------------------------------------
// Tensor Data Mover: stage a CHUNK x DIN f16 panel (row-major, stride DIN)
// from global memory into LDS. D# bit layout per CDNA5 ISA sec. 8.3/8.4:
//   group0: [1:0]=count=1, [63:32]=lds_addr, [120:64]=global_addr, type=2
//   group1: data_size=1(2B), tensor_dim0=DIN, tensor_dim1=NPTS,
//           tile_dim0=DIN, tile_dim1=CHUNK, tensor_dim0_stride=DIN
//   groups 2/3 (and trailing group on 6-arg toolchains): zero (2D tensor)
// ---------------------------------------------------------------------------
#if __has_builtin(__builtin_amdgcn_tensor_load_to_lds)
#define HAVE_TDM 1
typedef __attribute__((ext_vector_type(4))) unsigned int u32x4;
typedef __attribute__((ext_vector_type(8))) int i32x8;
typedef __attribute__((ext_vector_type(4))) int i32x4;

__device__ __forceinline__ void tdm_load_chunk(const _Float16* gptr,
                                               unsigned lds_byte) {
  unsigned long long ga = (unsigned long long)gptr;
  u32x4 g0;
  g0[0] = 1u;                                        // count=1, user descriptor
  g0[1] = lds_byte;                                  // lds_addr (bytes)
  g0[2] = (unsigned)(ga & 0xffffffffu);              // global_addr[31:0]
  g0[3] = (unsigned)((ga >> 32) & 0x1ffffffu) | (2u << 30);  // addr[56:32]|type=2
  i32x8 g1;
  g1[0] = (int)(1u << 16);                           // wg_mask=0, data_size=2B
  g1[1] = (int)((DIN & 0xffff) << 16);               // tensor_dim0[15:0]
  g1[2] = (int)((NPTS & 0xffff) << 16);              // tensor_dim1[15:0]
  g1[3] = (int)((DIN & 0xffff) << 16);               // tile_dim0
  g1[4] = (int)(CHUNK & 0xffff);                     // tile_dim1 (tile_dim2=0)
  g1[5] = (int)DIN;                                  // tensor_dim0_stride[31:0]
  g1[6] = 0;
  g1[7] = 0;
  i32x4 g2 = {0, 0, 0, 0};
  i32x4 g3 = {0, 0, 0, 0};
  i32x8 g4 = {0, 0, 0, 0, 0, 0, 0, 0};               // trailing group (clang-23)
  __builtin_amdgcn_tensor_load_to_lds(g0, g1, g2, g3, g4, 0);
}
#else
#define HAVE_TDM 0
#endif

// ---------------------------------------------------------------------------
// Kernel 2: nearest-neighbor distance via WMMA Gram matrix.
// Block = 8 waves = 8 row strips of one batch. The column panel is staged in
// LDS by the Tensor Data Mover (double-buffered, s_wait_tensorcnt), then each
// wave runs 16x16x32 f16 WMMAs against LDS-resident B tiles, fusing
// d2 = sq[n] - 2G (+sq[m] at the end), diagonal mask, and a running min
// reduced across lanes with shfl_xor.
// ---------------------------------------------------------------------------
__global__ __launch_bounds__(256) void k_knn(
    const _Float16* __restrict__ xh, const float* __restrict__ sq,
    float* __restrict__ nnd) {
  __shared__ _Float16 s_xh[2][CHUNK * DIN];         // 2 x 16 KB double buffer
  __shared__ float s_sq[NPTS];                      // 8 KB
  int tid  = threadIdx.x;
  int wave = tid >> 5;
  int lane = tid & 31;
  int tile = blockIdx.x * 8 + wave;                 // 0..1023
  int b    = tile >> 7;                             // uniform per block
  int rt   = tile & 127;
  int row_base = rt * 16;

  for (int i = tid; i < NPTS; i += 256) s_sq[i] = sq[b * NPTS + i];

  int m  = lane & 15;                               // A row index
  int h  = lane >> 4;                               // half-wave id
  int nn = lane & 15;                               // B column index

  // A tile: lane holds row (row_base+m), halfs K = h*8..h*8+7 and 16+h*8..+7
  union { v16h v; v8h p[2]; } A;
  const _Float16* arow = xh + ((size_t)b * NPTS + row_base + m) * DIN;
  A.p[0] = *(const v8h*)(arow + h * 8);
  A.p[1] = *(const v8h*)(arow + 16 + h * 8);

  float t[8];
#pragma unroll
  for (int j = 0; j < 8; ++j) t[j] = 3.0e38f;

#if HAVE_TDM
  if (wave == 0)
    tdm_load_chunk(xh + ((size_t)b * NPTS) * DIN,
                   (unsigned)(unsigned long long)&s_xh[0][0]);
#endif

  for (int chunk = 0; chunk < NCHUNK; ++chunk) {
#if HAVE_TDM
    if (wave == 0) {
      if (chunk + 1 < NCHUNK) {
        tdm_load_chunk(xh + ((size_t)b * NPTS + (chunk + 1) * CHUNK) * DIN,
                       (unsigned)(unsigned long long)&s_xh[(chunk + 1) & 1][0]);
        __builtin_amdgcn_s_wait_tensorcnt(1);       // chunk `chunk` resident
      } else {
        __builtin_amdgcn_s_wait_tensorcnt(0);
      }
    }
#else
    {
      const _Float16* src = xh + ((size_t)b * NPTS + chunk * CHUNK) * DIN;
      float4* dst = (float4*)&s_xh[chunk & 1][0];
      const float4* s4 = (const float4*)src;
      for (int i = tid; i < CHUNK * DIN * 2 / 16; i += 256) dst[i] = s4[i];
    }
#endif
    __syncthreads();                                // panel visible to all waves

    const _Float16* sb = &s_xh[chunk & 1][0];
    for (int ct = 0; ct < CHUNK / 16; ++ct) {
      // B tile from LDS: lane holds column, K = h*8..h*8+7 and 16+h*8..+7
      union { v16h v; v8h p[2]; } Bm;
      const _Float16* brow = sb + (ct * 16 + nn) * DIN;
      Bm.p[0] = *(const v8h*)(brow + h * 8);
      Bm.p[1] = *(const v8h*)(brow + 16 + h * 8);

      v8f c = {};
      c = __builtin_amdgcn_wmma_f32_16x16x32_f16(false, A.v, false, Bm.v,
                                                 (short)0, c, false, false);
      int col_base = chunk * CHUNK + ct * 16;
      float sc = s_sq[col_base + nn];
      bool diagTile = (col_base == row_base);
#pragma unroll
      for (int j = 0; j < 8; ++j) {
        // C layout: VGPR j, this lane -> row (j + 8*h), col nn
        float cand = sc - 2.0f * c[j];
        if (diagTile && ((j + 8 * h) == nn)) cand = 3.0e38f;
        t[j] = fminf(t[j], cand);
      }
    }
    __syncthreads();                                // release buffer for reuse
  }

  // min over the 16 lanes of each half-wave (columns) -> per-row min
#pragma unroll
  for (int j = 0; j < 8; ++j) {
    float v = t[j];
    v = fminf(v, __shfl_xor(v, 8, 32));
    v = fminf(v, __shfl_xor(v, 4, 32));
    v = fminf(v, __shfl_xor(v, 2, 32));
    v = fminf(v, __shfl_xor(v, 1, 32));
    t[j] = v;
  }
  if ((lane & 15) == 0) {
#pragma unroll
    for (int j = 0; j < 8; ++j) {
      int row = row_base + j + 8 * h;
      float d2 = s_sq[row] + t[j];
      nnd[b * NPTS + row] = sqrtf(fmaxf(d2, 0.f));
    }
  }
}

// ---------------------------------------------------------------------------
// Kernel 3: fused soft selector. Uses sim@y = fn * (fn^T y) (rank-1 collapse).
// One block per batch; 512 threads, 4 points each; LDS + shfl reductions.
// ---------------------------------------------------------------------------
__global__ __launch_bounds__(512) void k_select(
    const float* __restrict__ x, const float* __restrict__ sal,
    const float* __restrict__ nnd, const float* __restrict__ log_temp,
    float* __restrict__ fnbuf, float* __restrict__ ystar) {
  __shared__ float s_red[40];   // [0..36]=g, [37]=budget1, [38]=budget2
  int b = blockIdx.x, tid = threadIdx.x;
  if (tid < 40) s_red[tid] = 0.f;
  __syncthreads();

  float temp = expf(log_temp[0]);
  temp = fminf(fmaxf(temp, 0.1f), 10.0f);

  float y[4];
  float part = 0.f;
#pragma unroll
  for (int i = 0; i < 4; ++i) {
    int n = i * 512 + tid;
    const float* xr = x + ((size_t)b * NPTS + n) * DIN;
    float f[FDIM];
#pragma unroll
    for (int d = 0; d < DIN; ++d) f[d] = xr[d];
    float dd = nnd[b * NPTS + n];
    f[32] = dd; f[33] = dd; f[34] = dd; f[35] = 0.f;   // mean/min/max/std (k=1)
    float sv = sal[b * NPTS + n];
    f[36] = sv;
    float ss = 0.f;
#pragma unroll
    for (int c = 0; c < FDIM; ++c) ss += f[c] * f[c];
    float inv = 1.0f / (sqrtf(ss) + 1e-8f);
    float* fo = fnbuf + ((size_t)b * NPTS + n) * FDIM;
#pragma unroll
    for (int c = 0; c < FDIM; ++c) fo[c] = f[c] * inv;
    float arg = (sv / (2.0f * 0.5f) - 0.5f) / temp;    // LAM = 0.5
    float yv = 1.0f / (1.0f + expf(-arg));
    y[i] = yv;
    part += yv;
  }
  for (int msk = 16; msk >= 1; msk >>= 1) part += __shfl_xor(part, msk, 32);
  if ((tid & 31) == 0) atomicAdd(&s_red[37], part);
  __syncthreads();

  float budget = fmaxf(s_red[37], 1e-6f);
  float sc1 = fminf(KSEL / budget, 1.0f);

  float gp[FDIM];
#pragma unroll
  for (int c = 0; c < FDIM; ++c) gp[c] = 0.f;
#pragma unroll
  for (int i = 0; i < 4; ++i) {
    int n = i * 512 + tid;
    y[i] *= sc1;
    const float* fo = fnbuf + ((size_t)b * NPTS + n) * FDIM;
#pragma unroll
    for (int c = 0; c < FDIM; ++c) gp[c] += y[i] * fo[c];
  }
#pragma unroll
  for (int c = 0; c < FDIM; ++c) {
    float v = gp[c];
    for (int msk = 16; msk >= 1; msk >>= 1) v += __shfl_xor(v, msk, 32);
    if ((tid & 31) == 0) atomicAdd(&s_red[c], v);
  }
  __syncthreads();

  float y2[4];
  float part2 = 0.f;
#pragma unroll
  for (int i = 0; i < 4; ++i) {
    int n = i * 512 + tid;
    const float* fo = fnbuf + ((size_t)b * NPTS + n) * FDIM;
    float ov = 0.f;
#pragma unroll
    for (int c = 0; c < FDIM; ++c) ov += fo[c] * s_red[c];
    y2[i] = y[i] / (1.0f + ov);
    part2 += y2[i];
  }
  for (int msk = 16; msk >= 1; msk >>= 1) part2 += __shfl_xor(part2, msk, 32);
  if ((tid & 31) == 0) atomicAdd(&s_red[38], part2);
  __syncthreads();

  float b2 = fmaxf(s_red[38], 1e-6f);
  float sc2 = fminf(KSEL / b2, 1.0f);
#pragma unroll
  for (int i = 0; i < 4; ++i) {
    int n = i * 512 + tid;
    ystar[b * NPTS + n] = y2[i] * sc2;
  }
}

// ---------------------------------------------------------------------------
// Kernel 4: lifted = tanh(((dense - mu)/sigma) @ Wl + bl), dense = [x, d,d,d,0]
// ---------------------------------------------------------------------------
__global__ __launch_bounds__(256) void k_lift(
    const float* __restrict__ x, const float* __restrict__ nnd,
    const float* __restrict__ mu, const float* __restrict__ sigma,
    const float* __restrict__ Wl, const float* __restrict__ bl,
    float* __restrict__ lifted) {
  __shared__ float sW[ADIM * KLIFT];
  __shared__ float smu[ADIM], ssg[ADIM], sbl[KLIFT];
  int tid = threadIdx.x;
  for (int i = tid; i < ADIM * KLIFT; i += 256) sW[i] = Wl[i];
  if (tid < ADIM) { smu[tid] = mu[tid]; ssg[tid] = sigma[tid]; }
  if (tid < KLIFT) sbl[tid] = bl[tid];
  __syncthreads();

  int n = blockIdx.x * 256 + tid;                 // global point id
  const float* xr = x + (size_t)n * DIN;
  float z[ADIM];
#pragma unroll
  for (int d = 0; d < DIN; ++d) z[d] = xr[d];
  float dd = nnd[n];
  z[32] = dd; z[33] = dd; z[34] = dd; z[35] = 0.f;
#pragma unroll
  for (int a = 0; a < ADIM; ++a) z[a] = (z[a] - smu[a]) / ssg[a];
  float* lo = lifted + (size_t)n * KLIFT;
#pragma unroll
  for (int k = 0; k < KLIFT; ++k) {
    float acc = sbl[k];
#pragma unroll
    for (int a = 0; a < ADIM; ++a) acc += z[a] * sW[a * KLIFT + k];
    lo[k] = tanhf(acc);
  }
}

// ---------------------------------------------------------------------------
// Kernel 5: top-16 of y_star per batch (iterative argmax, ties -> lower idx),
// then tokens = cloud @ Wp + bp.
// ---------------------------------------------------------------------------
__global__ __launch_bounds__(256) void k_topk_tokens(
    const float* __restrict__ ystar, const float* __restrict__ lifted,
    const float* __restrict__ Wp, const float* __restrict__ bp,
    float* __restrict__ tokens) {
  __shared__ float yv[NPTS];
  __shared__ float rv[256];
  __shared__ int   ri[256];
  __shared__ int   topidx[KEFF];
  __shared__ float cloud[KEFF * KLIFT];
  int b = blockIdx.x, tid = threadIdx.x;
  for (int i = tid; i < NPTS; i += 256) yv[i] = ystar[b * NPTS + i];
  __syncthreads();

  for (int r = 0; r < KEFF; ++r) {
    float bestv = -3.4e38f; int besti = NPTS;
#pragma unroll
    for (int j = 0; j < NPTS / 256; ++j) {
      int n = tid + j * 256;
      float v = yv[n];
      if (v > bestv || (v == bestv && n < besti)) { bestv = v; besti = n; }
    }
    rv[tid] = bestv; ri[tid] = besti;
    __syncthreads();
    for (int s = 128; s >= 1; s >>= 1) {
      if (tid < s) {
        float v2 = rv[tid + s]; int i2 = ri[tid + s];
        if (v2 > rv[tid] || (v2 == rv[tid] && i2 < ri[tid])) {
          rv[tid] = v2; ri[tid] = i2;
        }
      }
      __syncthreads();
    }
    if (tid == 0) { topidx[r] = ri[0]; yv[ri[0]] = -3.4e38f; }
    __syncthreads();
  }

  {
    int k = tid >> 4, c = tid & 15;               // 256 threads -> 16x16 cloud
    cloud[tid] = lifted[((size_t)b * NPTS + topidx[k]) * KLIFT + c];
  }
  __syncthreads();

  int d = tid;                                    // output column 0..255
#pragma unroll
  for (int k = 0; k < KEFF; ++k) {
    float acc = bp[d];
#pragma unroll
    for (int c = 0; c < KLIFT; ++c) acc += cloud[k * KLIFT + c] * Wp[c * DMODEL + d];
    tokens[((size_t)b * KEFF + k) * DMODEL + d] = acc;
  }
}

// ---------------------------------------------------------------------------
extern "C" void kernel_launch(void* const* d_in, const int* in_sizes, int n_in,
                              void* d_out, int out_size, void* d_ws, size_t ws_size,
                              hipStream_t stream) {
  const float* x        = (const float*)d_in[0];
  const float* log_temp = (const float*)d_in[1];
  const float* W1       = (const float*)d_in[2];
  const float* b1       = (const float*)d_in[3];
  const float* W2       = (const float*)d_in[4];
  const float* b2       = (const float*)d_in[5];
  const float* mu       = (const float*)d_in[6];
  const float* sigma    = (const float*)d_in[7];
  const float* Wl       = (const float*)d_in[8];
  const float* bl       = (const float*)d_in[9];
  const float* Wp       = (const float*)d_in[10];
  const float* bp       = (const float*)d_in[11];

  char* ws = (char*)d_ws;
  size_t off = 0;
  _Float16* xh   = (_Float16*)(ws + off); off += (size_t)BATCH * NPTS * DIN * 2;   // 1 MB
  float* sq      = (float*)(ws + off);    off += (size_t)BATCH * NPTS * 4;          // 64 KB
  float* sal     = (float*)(ws + off);    off += (size_t)BATCH * NPTS * 4;          // 64 KB
  float* nnd     = (float*)(ws + off);    off += (size_t)BATCH * NPTS * 4;          // 64 KB
  float* fnbuf   = (float*)(ws + off);    off += (size_t)BATCH * NPTS * FDIM * 4;   // 2.4 MB
  float* lifted  = (float*)(ws + off);    off += (size_t)BATCH * NPTS * KLIFT * 4;  // 1 MB
  (void)ws_size; (void)n_in; (void)in_sizes; (void)out_size;

  float* tokens = (float*)d_out;                            // [B,16,256]
  float* ystar  = (float*)d_out + (size_t)BATCH * KEFF * DMODEL;  // [B,N]

  // 1. saliency MLP + f16 conversion + |x|^2
  k_saliency<<<(BATCH * NPTS) / 256, 256, 0, stream>>>(x, W1, b1, W2, b2,
                                                       xh, sq, sal);
  // 2. nearest-neighbor distance (TDM-staged LDS + WMMA Gram matrix)
  k_knn<<<(BATCH * NPTS / 16) / 8, 256, 0, stream>>>(xh, sq, nnd);
  // 3. soft selector with rank-1 similarity collapse
  k_select<<<BATCH, 512, 0, stream>>>(x, sal, nnd, log_temp, fnbuf, ystar);
  // 4. lifting MLP
  k_lift<<<(BATCH * NPTS) / 256, 256, 0, stream>>>(x, nnd, mu, sigma, Wl, bl,
                                                   lifted);
  // 5. top-16 + token projection
  k_topk_tokens<<<BATCH, 256, 0, stream>>>(ystar, lifted, Wp, bp, tokens);
}
